// AnswerSynthesisDecoder_12987981103236
// MI455X (gfx1250) — matrix operand order repaired and checked by reading, not compiled, not run on hardware
//
#include <hip/hip_runtime.h>
#include <hip/hip_bf16.h>
#include <math.h>

typedef __attribute__((ext_vector_type(8)))  __bf16 v8bf;
typedef __attribute__((ext_vector_type(16))) __bf16 v16bf;
typedef __attribute__((ext_vector_type(8)))  float  v8f;

#define D2 1024
#define D4 2048
#define E_ 1024
#define H_ 512
#define V_ 32000
#define L_ 64
#define G3 6144   // 3*D4

// ---------- wave32 helpers ----------
__device__ __forceinline__ float wave_red(float v) {
#pragma unroll
  for (int o = 16; o > 0; o >>= 1) v += __shfl_xor(v, o, 32);
  return v;
}
// K must be a multiple of 128; 16B-aligned rows. One b128 load per operand per iter.
__device__ __forceinline__ float wave_dot4(const float* __restrict__ a,
                                           const float* __restrict__ b,
                                           int K, int lane) {
  float s = 0.f;
  for (int k = lane * 4; k < K; k += 128) {
    const float4 av = *reinterpret_cast<const float4*>(a + k);
    const float4 bv = *reinterpret_cast<const float4*>(b + k);
    s = fmaf(av.x, bv.x, s); s = fmaf(av.y, bv.y, s);
    s = fmaf(av.z, bv.z, s); s = fmaf(av.w, bv.w, s);
  }
  return wave_red(s);
}

// ---------- bf16 weight packing ----------
// out[m*K + k] = (bf16) A[m*lda + col_off + k]; one block per row m.
__global__ void k_pack_bf16(const float* __restrict__ A, int lda, int col_off,
                            __bf16* __restrict__ out, int K) {
  size_t m = blockIdx.x;
  const float* src = A + m * (size_t)lda + col_off;
  __bf16* dst = out + m * (size_t)K;
  for (int k = threadIdx.x; k < K; k += blockDim.x) dst[k] = (__bf16)src[k];
}

// ---------- phase A: pooled h (two-stage deterministic column sum) ----------
__global__ void k_colsum_part(const float* __restrict__ hq, const float* __restrict__ hp,
                              float* __restrict__ part) {
  int b = blockIdx.x;                 // 32 blocks
  int j4 = threadIdx.x * 4;           // 256 threads * 4 cols = 1024
  float4 s = {0.f, 0.f, 0.f, 0.f};
  for (int n = b * 128; n < (b + 1) * 128; ++n) {
    float4 v = *reinterpret_cast<const float4*>(hp + (size_t)n * D2 + j4);
    s.x += v.x; s.y += v.y; s.z += v.z; s.w += v.w;
  }
  if (b == 0) {
    for (int m = 0; m < 32; ++m) {
      float4 v = *reinterpret_cast<const float4*>(hq + (size_t)m * D2 + j4);
      s.x += v.x; s.y += v.y; s.z += v.z; s.w += v.w;
    }
  }
  *reinterpret_cast<float4*>(part + (size_t)b * D2 + j4) = s;
}
__global__ void k_colsum_red(const float* __restrict__ part, float* __restrict__ h) {
  int j = blockIdx.x * blockDim.x + threadIdx.x;
  if (j >= D2) return;
  float s = 0.f;
  for (int b = 0; b < 32; ++b) s += part[(size_t)b * D2 + j];
  h[j] = s;
}

// u_a_h = u_a_w @ h ; d0 rows = tanh(w_d @ {h_q0,h_p0} + b)
__global__ void k_init_rows(const float* __restrict__ uaw, const float* __restrict__ wdw,
                            const float* __restrict__ wdb,
                            const float* __restrict__ hq0, const float* __restrict__ hp0,
                            const float* __restrict__ h,
                            float* __restrict__ uah, float* __restrict__ dA) {
  int wid  = (blockIdx.x * blockDim.x + threadIdx.x) >> 5;
  int lane = threadIdx.x & 31;
  if (wid < D2) {
    float s = wave_dot4(uaw + (size_t)wid * D2, h, D2, lane);
    if (!lane) uah[wid] = s;
  } else if (wid < 2 * D2) {
    int r = wid - D2;
    float s = wave_dot4(wdw + (size_t)r * D2, hq0, D2, lane);
    if (!lane) dA[r] = tanhf(s + wdb[r]);
  } else if (wid < 3 * D2) {
    int r = wid - 2 * D2;
    float s = wave_dot4(wdw + (size_t)r * D2, hp0, D2, lane);
    if (!lane) dA[D2 + r] = tanhf(s + wdb[r]);
  }
}

// ---------- WMMA GEMM on pre-packed bf16: C(M x 64) = A(M x K) @ B(64 x K)^T ----------
// transC=1 stores C[n*ldc + m]. K multiple of 64.
__global__ void k_wmma_gemm_bf16(const __bf16* __restrict__ A,
                                 const __bf16* __restrict__ B,
                                 const float* __restrict__ bias,
                                 float* __restrict__ C, int ldc, int transC,
                                 int Mtot, int K) {
  int wid  = (blockIdx.x * blockDim.x + threadIdx.x) >> 5;
  int lane = threadIdx.x & 31;
  int tile_n = wid & 3;        // 64/16 = 4 n-tiles
  int tile_m = wid >> 2;
  if (tile_m >= Mtot / 16) return;
  int half = lane >> 4;
  int l16  = lane & 15;
  const __bf16* Arow = A + ((size_t)(tile_m * 16 + l16)) * K;
  const __bf16* Brow = B + ((size_t)(tile_n * 16 + l16)) * K;
  int hb = half * 8;
  v8f acc = {};
  for (int kb = 0; kb < K; kb += 64) {
#pragma unroll
    for (int u = 0; u < 2; ++u) {
      int k0 = kb + u * 32;
      // A 16x32 fragment: two contiguous 8-elem (16B) runs per lane
      v8bf alo = *reinterpret_cast<const v8bf*>(Arow + k0 + hb);
      v8bf ahi = *reinterpret_cast<const v8bf*>(Arow + k0 + hb + 16);
      v16bf a = __builtin_shufflevector(alo, ahi, 0, 1, 2, 3, 4, 5, 6, 7,
                                        8, 9, 10, 11, 12, 13, 14, 15);
      // B 32x16 fragment: two contiguous 8-elem (16B) runs per lane
      v8bf blo = *reinterpret_cast<const v8bf*>(Brow + k0 + hb);
      v8bf bhi = *reinterpret_cast<const v8bf*>(Brow + k0 + hb + 16);
      v16bf b = __builtin_shufflevector(blo, bhi, 0, 1, 2, 3, 4, 5, 6, 7,
                                        8, 9, 10, 11, 12, 13, 14, 15);
      acc = __builtin_amdgcn_wmma_f32_16x16x32_bf16(false, a, false, b,
                                                    (short)0, acc, false, false);
    }
  }
#pragma unroll
  for (int v = 0; v < 8; ++v) {
    int m = tile_m * 16 + v + half * 8;
    int n = tile_n * 16 + l16;
    float val = acc[v];
    if (bias) val += bias[m];
    if (transC) C[(size_t)n * ldc + m] = val;
    else        C[(size_t)m * ldc + n] = val;
  }
}

// ---------- sequential per-step kernels ----------
__global__ void k_att_mv(const float* __restrict__ waw, const float* __restrict__ vw,
                         const float* __restrict__ uah, const float* __restrict__ d,
                         float* __restrict__ tatt) {
  int wid  = (blockIdx.x * blockDim.x + threadIdx.x) >> 5;
  int lane = threadIdx.x & 31;
  if (wid >= 2 * D2) return;
  int row = wid >> 10;
  int r   = wid & (D2 - 1);
  float s = wave_dot4(waw + (size_t)r * D2, d + row * D2, D2, lane);
  if (!lane) tatt[wid] = vw[r] * tanhf(s + uah[r]);
}

__global__ void k_att_reduce(const float* __restrict__ tatt, const float* __restrict__ d,
                             float* __restrict__ c) {
  __shared__ float s0s[1024], s1s[1024];
  __shared__ float a0_s, a1_s;
  int j = threadIdx.x;
  s0s[j] = tatt[j];
  s1s[j] = tatt[D2 + j];
  __syncthreads();
  for (int o = 512; o > 0; o >>= 1) {
    if (j < o) { s0s[j] += s0s[j + o]; s1s[j] += s1s[j + o]; }
    __syncthreads();
  }
  if (j == 0) {
    float s0 = s0s[0], s1 = s1s[0];
    float m = fmaxf(s0, s1);
    float e0 = expf(s0 - m), e1 = expf(s1 - m);
    float inv = 1.f / (e0 + e1);
    a0_s = e0 * inv; a1_s = e1 * inv;
  }
  __syncthreads();
  c[j] = a0_s * d[j] + a1_s * d[D2 + j];
}

__global__ void k_gate_mv(const float* __restrict__ wih, const float* __restrict__ whh,
                          const float* __restrict__ bhh, const float* __restrict__ GIW, int t,
                          const float* __restrict__ d, const float* __restrict__ c,
                          float* __restrict__ ghv, float* __restrict__ giv) {
  int wid  = (blockIdx.x * blockDim.x + threadIdx.x) >> 5;
  int lane = threadIdx.x & 31;
  if (wid >= G3) return;
  float gh = wave_dot4(whh + (size_t)wid * D4, d, D4, lane);
  float gi = wave_dot4(wih + (size_t)wid * D4 + E_, c, D2, lane);
  if (!lane) {
    ghv[wid] = gh + bhh[wid];
    giv[wid] = gi + GIW[(size_t)wid * L_ + t];
  }
}

__global__ void k_gru(const float* __restrict__ giv, const float* __restrict__ ghv,
                      const float* __restrict__ d, float* __restrict__ dn) {
  int j = blockIdx.x * blockDim.x + threadIdx.x;
  if (j >= D4) return;
  float r = 1.f / (1.f + expf(-(giv[j] + ghv[j])));
  float z = 1.f / (1.f + expf(-(giv[D4 + j] + ghv[D4 + j])));
  float n = tanhf(giv[2 * D4 + j] + r * ghv[2 * D4 + j]);
  dn[j] = (1.f - z) * n + z * d[j];
}

// r_t rows i and i+512 fused with maxout; writes bf16 row for the vocab GEMM
__global__ void k_rt_max(const float* __restrict__ urw, const float* __restrict__ vrw,
                         const float* __restrict__ RW, int t,
                         const float* __restrict__ c, const float* __restrict__ hn,
                         __bf16* __restrict__ Rbf) {
  int wid  = (blockIdx.x * blockDim.x + threadIdx.x) >> 5;
  int lane = threadIdx.x & 31;
  if (wid >= H_) return;
  float a = RW[(size_t)wid * L_ + t]
          + wave_dot4(urw + (size_t)wid * D2, c, D2, lane)
          + wave_dot4(vrw + (size_t)wid * D4, hn, D4, lane);
  int r2 = wid + H_;
  float b = RW[(size_t)r2 * L_ + t]
          + wave_dot4(urw + (size_t)r2 * D2, c, D2, lane)
          + wave_dot4(vrw + (size_t)r2 * D4, hn, D4, lane);
  if (!lane) Rbf[(size_t)t * H_ + wid] = (__bf16)fmaxf(a, b);
}

// ---------- phase C: softmax in place over each row of (64, 32000) ----------
__global__ void k_softmax(float* __restrict__ out) {
  __shared__ float red[256];
  int t = blockIdx.x;
  float* p = out + (size_t)t * V_;
  int tid = threadIdx.x;
  float mx = -3.4e38f;
  for (int v = tid; v < V_; v += 256) mx = fmaxf(mx, p[v]);
  red[tid] = mx; __syncthreads();
  for (int o = 128; o > 0; o >>= 1) { if (tid < o) red[tid] = fmaxf(red[tid], red[tid + o]); __syncthreads(); }
  mx = red[0];
  __syncthreads();
  float sum = 0.f;
  for (int v = tid; v < V_; v += 256) { float e = expf(p[v] - mx); p[v] = e; sum += e; }
  red[tid] = sum; __syncthreads();
  for (int o = 128; o > 0; o >>= 1) { if (tid < o) red[tid] += red[tid + o]; __syncthreads(); }
  float inv = 1.f / red[0];
  __syncthreads();
  for (int v = tid; v < V_; v += 256) p[v] *= inv;
}

extern "C" void kernel_launch(void* const* d_in, const int* in_sizes, int n_in,
                              void* d_out, int out_size, void* d_ws, size_t ws_size,
                              hipStream_t stream) {
  const float* hq  = (const float*)d_in[0];
  const float* hp  = (const float*)d_in[1];
  const float* ans = (const float*)d_in[2];
  const float* vw  = (const float*)d_in[3];
  const float* wdw = (const float*)d_in[4];
  const float* wdb = (const float*)d_in[5];
  const float* waw = (const float*)d_in[6];
  const float* uaw = (const float*)d_in[7];
  const float* wrw = (const float*)d_in[8];
  const float* urw = (const float*)d_in[9];
  const float* vrw = (const float*)d_in[10];
  const float* wow = (const float*)d_in[11];
  const float* wih = (const float*)d_in[12];
  const float* whh = (const float*)d_in[13];
  const float* bih = (const float*)d_in[14];
  const float* bhh = (const float*)d_in[15];
  float* out = (float*)d_out;

  // ---- float workspace ----
  float* F = (float*)d_ws;
  float* h    = F;             // 1024
  float* uah  = F + 1024;      // 1024
  float* dA   = F + 2048;      // 2048
  float* dB   = F + 4096;      // 2048
  float* c    = F + 6144;      // 1024
  float* tatt = F + 7168;      // 2048
  float* ghv  = F + 9216;      // 6144
  float* giv  = F + 15360;     // 6144
  float* part = F + 21504;     // 32*1024
  float* GIW  = F + 54272;     // 6144*64
  float* RW   = F + 447488;    // 1024*64  -> float end = 513024 (32B aligned)

  // ---- bf16 workspace (after float region) ----
  __bf16* B16  = (__bf16*)(F + 513024);
  __bf16* ansb = B16;                 // 64*1024
  __bf16* Rbf  = B16 + 65536;         // 64*512
  __bf16* wihb = B16 + 98304;         // 6144*1024
  __bf16* wrwb = B16 + 6389760;       // 1024*1024
  __bf16* wowb = B16 + 7438336;       // 32000*512  -> ~49.7 MB total ws

  // Phase A: pooled h, u_a_h, d0
  k_colsum_part<<<32, 256, 0, stream>>>(hq, hp, part);
  k_colsum_red<<<4, 256, 0, stream>>>(part, h);
  k_init_rows<<<384, 256, 0, stream>>>(uaw, wdw, wdb, hq, hp, h, uah, dA);

  // Pack weights/activations to bf16 once per call
  k_pack_bf16<<<L_, 256, 0, stream>>>(ans, E_, 0, ansb, E_);
  k_pack_bf16<<<G3, 256, 0, stream>>>(wih, D4, 0, wihb, E_);
  k_pack_bf16<<<D2, 256, 0, stream>>>(wrw, D2, 0, wrwb, D2);
  k_pack_bf16<<<V_, 256, 0, stream>>>(wow, H_, 0, wowb, H_);

  // Phase A (WMMA): GIW = w_ih[:, :E] @ ans^T + b_ih ; RW = w_r @ ans^T
  k_wmma_gemm_bf16<<<192, 256, 0, stream>>>(wihb, ansb, bih, GIW, L_, 0, G3, E_);
  k_wmma_gemm_bf16<<<32, 256, 0, stream>>>(wrwb, ansb, nullptr, RW, L_, 0, D2, E_);

  // Phase B: 64 sequential steps (weights stay L2-resident across steps)
  float* dcur = dA; float* dnext = dB;
  for (int t = 0; t < L_; ++t) {
    k_att_mv<<<256, 256, 0, stream>>>(waw, vw, uah, dcur, tatt);
    k_att_reduce<<<1, 1024, 0, stream>>>(tatt, dcur, c);
    k_gate_mv<<<768, 256, 0, stream>>>(wih, whh, bhh, GIW, t, dcur, c, ghv, giv);
    k_gru<<<8, 256, 0, stream>>>(giv, ghv, dcur, dnext);
    k_rt_max<<<64, 256, 0, stream>>>(urw, vrw, RW, t, c, dnext, Rbf);
    float* tmp = dcur; dcur = dnext; dnext = tmp;
  }

  // Phase C (WMMA): logits (32000 x 64) stored transposed into out (64, 32000), then softmax
  k_wmma_gemm_bf16<<<1000, 256, 0, stream>>>(wowb, Rbf, nullptr, out, V_, 1, V_, H_);
  k_softmax<<<64, 256, 0, stream>>>(out);
}
